// EdgeBlock_44865228374366
// MI455X (gfx1250) — matrix operand order repaired and checked by reading, not compiled, not run on hardware
//
#include <hip/hip_runtime.h>
#include <hip/hip_bf16.h>
#include <stdint.h>

// E=640000 edges, D=128, concat K1=512, hidden H=256, out O=128.
// Layer1: [E,512] @ [512,256] (+b1, relu) ; Layer2: [E,256] @ [256,128] (+b2)
// f16 WMMA (v_wmma_f32_16x16x32_f16), f32 accum.
//   - Block = 256 threads = 8 waves, 128 edges per block (16 edges per wave).
//   - Weights pre-packed to f16 B-fragment layout in d_ws (L2-resident).
//   - Weight k-slices staged into LDS with CDNA5 async-to-LDS DMA in SADDR
//     form (uniform slice base in SGPRs, one constant per-lane offset VGPR,
//     chunk selection via 24-bit inst offset) -> no per-slice VGPR address
//     arithmetic, no spills.
//   - Double-buffered staging overlaps DMA with the WMMA inner loop.
//   - A fragments software-pipelined one k-step ahead from global f32;
//     custom asm barrier waits only ASYNCcnt/DScnt so those loads stay in
//     flight across k-step boundaries (plain __syncthreads drains LOADcnt).
//   - All LDS reads/writes off a single per-lane base + immediate offsets.

typedef __attribute__((ext_vector_type(16))) _Float16 v16h;
typedef __attribute__((ext_vector_type(8)))  float    v8f;

union Frag {
    v16h     v;
    uint32_t u[8];
    uint4    q[2];
};

#define DIMD   128
#define DIMH   256
#define DIMO   128
#define MTILE  128
#define NWAVES 8

// CDNA5 async global->LDS DMA, SADDR form: 64B per lane (4 x b128 chunks).
// INST_OFFSET is added to both the LDS and global addresses (ISA 15.18),
// and our LDS/global chunk offsets advance identically, so chunks use imm.
__device__ __forceinline__ void async_slice64(uint32_t lds_dst, uint32_t voff,
                                              const void* sbase) {
    asm volatile("global_load_async_to_lds_b128 %0, %1, %2\n\t"
                 "global_load_async_to_lds_b128 %0, %1, %2 offset:16\n\t"
                 "global_load_async_to_lds_b128 %0, %1, %2 offset:32\n\t"
                 "global_load_async_to_lds_b128 %0, %1, %2 offset:48"
                 :: "v"(lds_dst), "v"(voff), "s"(sbase));
}
// 32B per lane (2 x b128 chunks).
__device__ __forceinline__ void async_slice32(uint32_t lds_dst, uint32_t voff,
                                              const void* sbase) {
    asm volatile("global_load_async_to_lds_b128 %0, %1, %2\n\t"
                 "global_load_async_to_lds_b128 %0, %1, %2 offset:16"
                 :: "v"(lds_dst), "v"(voff), "s"(sbase));
}

// Workgroup barrier that drains only ASYNCcnt + DScnt (NOT LOADcnt), so
// pipelined global loads stay in flight across the barrier.
__device__ __forceinline__ void wg_barrier_async_ds() {
    asm volatile("s_wait_asynccnt 0x0\n\t"
                 "s_wait_dscnt 0x0\n\t"
                 "s_barrier_signal -1\n\t"
                 "s_barrier_wait -1" ::: "memory");
}
__device__ __forceinline__ void wg_barrier_ds() {
    asm volatile("s_wait_dscnt 0x0\n\t"
                 "s_barrier_signal -1\n\t"
                 "s_barrier_wait -1" ::: "memory");
}

// ---------------------------------------------------------------------------
// Pack W1 [512,256] f32 and W2 [256,128] f32 into f16 B-fragment layout.
// B fragment (32x16, KxN): lanes 0-15 hold column N=lane, vgpr v = K pair
// (2v,2v+1) for v<4, (8+2v,9+2v) for v>=4; lanes 16-31 hold N=lane-16, same +8.
// Storage: frag[ks][nt] is 256 dwords; within frag: lane*8 + v dwords.
// ---------------------------------------------------------------------------
__global__ void pack_weights_kernel(const float* __restrict__ W1,
                                    const float* __restrict__ W2,
                                    uint32_t* __restrict__ W1p,
                                    uint32_t* __restrict__ W2p) {
    int idx = blockIdx.x * blockDim.x + threadIdx.x;
    if (idx < 65536) {
        int v    = idx & 7;
        int lane = (idx >> 3) & 31;
        int nt   = (idx >> 8) & 15;
        int ks   = idx >> 12;
        int n = nt * 16 + (lane & 15);
        int k = ks * 32 + ((lane & 16) ? 8 : 0) + v * 2 + ((v & 4) ? 8 : 0);
        W1p[idx] = __builtin_bit_cast(unsigned int,
            __builtin_amdgcn_cvt_pkrtz(W1[k * DIMH + n], W1[(k + 1) * DIMH + n]));
    } else if (idx < 65536 + 16384) {
        int j    = idx - 65536;
        int v    = j & 7;
        int lane = (j >> 3) & 31;
        int nt   = (j >> 8) & 7;
        int ks   = j >> 11;
        int n = nt * 16 + (lane & 15);
        int k = ks * 32 + ((lane & 16) ? 8 : 0) + v * 2 + ((v & 4) ? 8 : 0);
        W2p[j] = __builtin_bit_cast(unsigned int,
            __builtin_amdgcn_cvt_pkrtz(W2[k * DIMO + n], W2[(k + 1) * DIMO + n]));
    }
}

// ---------------------------------------------------------------------------
// Fused edge-MLP kernel.
// ---------------------------------------------------------------------------
__global__ __launch_bounds__(256, 1)
void edge_mlp_kernel(const float* __restrict__ edge_attr,
                     const float* __restrict__ node_attr,
                     const float* __restrict__ global_attr,
                     const int*   __restrict__ senders,
                     const int*   __restrict__ receivers,
                     const uint32_t* __restrict__ W1p,
                     const float* __restrict__ b1,
                     const uint32_t* __restrict__ W2p,
                     const float* __restrict__ b2,
                     float* __restrict__ out) {
    // Per-wave h staging: 8 waves * 4096 halves = 64KB
    __shared__ __align__(16) _Float16 hbuf[NWAVES * 4096];
    // Double-buffered weight slice staging: 2 * 16KB, contiguous.
    __shared__ __align__(16) uint32_t wstage[2][4096];

    const int tid  = threadIdx.x;
    const int wave = tid >> 5;
    const int lane = tid & 31;
    const int l16  = lane & 15;
    const int hi   = lane >> 4;              // 0 for lanes 0-15, 1 for 16-31

    const int m = blockIdx.x * MTILE + wave * 16 + l16;  // this lane's edge row

    const int recv = receivers[m];
    const int send = senders[m];
    const float* rows[4];
    rows[0] = edge_attr + (size_t)m * DIMD;
    rows[1] = node_attr + (size_t)recv * DIMD;
    rows[2] = node_attr + (size_t)send * DIMD;
    rows[3] = global_attr;

    // Single per-lane LDS base; buffer parity + fragment index are immediates.
    const uint4* wbase = (const uint4*)(&wstage[0][0]) + lane * 2;

    // Constant per-lane async operands (raw LDS offsets + per-lane global off).
    const uint32_t ws0   = (uint32_t)(uintptr_t)&wstage[0][0];
    const uint32_t ldst0 = ws0 + tid * 64;          // layer-1 dst, buffer 0
    const uint32_t ldst1 = ldst0 + 16384;           // layer-1 dst, buffer 1
    const uint32_t l2d0  = ws0 + tid * 32;          // layer-2 dst, buffer 0
    const uint32_t l2d1  = l2d0 + 16384;            // layer-2 dst, buffer 1
    const uint32_t voff64 = (uint32_t)tid * 64;     // per-lane global offset
    const uint32_t voff32 = (uint32_t)tid * 32;

    // ---- Prologue: DMA W1 slice 0 into wstage[0]; preload A regs for ks=0.
    async_slice64(ldst0, voff64, W1p);
    float4 a0, a1, a2, a3;
    {
        const float* base = rows[0] + hi * 8;
        a0 = *(const float4*)(base + 0);
        a1 = *(const float4*)(base + 4);
        a2 = *(const float4*)(base + 16);
        a3 = *(const float4*)(base + 20);
    }
    wg_barrier_async_ds();

    // ---------------- Layer 1: [16,512] x [512,256] -> [16,256] ----------------
    v8f acc[16];
#pragma unroll
    for (int i = 0; i < 16; ++i) acc[i] = (v8f){0.f,0.f,0.f,0.f,0.f,0.f,0.f,0.f};

#pragma unroll
    for (int ks = 0; ks < 16; ++ks) {
        // Build A fragment (16x32 f16) from the pipelined registers.
        // lane<16: K offsets 0..7 (v0..3), 16..23 (v4..7); lane>=16: +8.
        Frag A;
        A.u[0] = __builtin_bit_cast(unsigned int, __builtin_amdgcn_cvt_pkrtz(a0.x, a0.y));
        A.u[1] = __builtin_bit_cast(unsigned int, __builtin_amdgcn_cvt_pkrtz(a0.z, a0.w));
        A.u[2] = __builtin_bit_cast(unsigned int, __builtin_amdgcn_cvt_pkrtz(a1.x, a1.y));
        A.u[3] = __builtin_bit_cast(unsigned int, __builtin_amdgcn_cvt_pkrtz(a1.z, a1.w));
        A.u[4] = __builtin_bit_cast(unsigned int, __builtin_amdgcn_cvt_pkrtz(a2.x, a2.y));
        A.u[5] = __builtin_bit_cast(unsigned int, __builtin_amdgcn_cvt_pkrtz(a2.z, a2.w));
        A.u[6] = __builtin_bit_cast(unsigned int, __builtin_amdgcn_cvt_pkrtz(a3.x, a3.y));
        A.u[7] = __builtin_bit_cast(unsigned int, __builtin_amdgcn_cvt_pkrtz(a3.z, a3.w));

        // Prefetch A registers for the next k-step (stays in flight across
        // the custom barrier; consumed by next iteration's cvt).
        if (ks < 15) {
            const float* base = rows[(ks + 1) >> 2] + ((ks + 1) & 3) * 32 + hi * 8;
            a0 = *(const float4*)(base + 0);
            a1 = *(const float4*)(base + 4);
            a2 = *(const float4*)(base + 16);
            a3 = *(const float4*)(base + 20);
        }

        // Kick off async DMA of the next weight slice (uniform SGPR base).
        if (ks < 15) {
            async_slice64((ks & 1) ? ldst0 : ldst1, voff64, W1p + (ks + 1) * 4096);
        } else {
            // Prefetch W2 slice 0 into wstage[0] (last read at ks=14).
            async_slice32(l2d0, voff32, W2p);
        }

#pragma unroll
        for (int nt = 0; nt < 16; ++nt) {
            Frag B;
            B.q[0] = wbase[(ks & 1) * 1024 + nt * 64 + 0];  // ds_load_b128 imm offs
            B.q[1] = wbase[(ks & 1) * 1024 + nt * 64 + 1];
            acc[nt] = __builtin_amdgcn_wmma_f32_16x16x32_f16(
                false, A.v, false, B.v, (short)0, acc[nt], false, false);
        }
        wg_barrier_async_ds();
    }

    // Bias + ReLU + f16, scatter C-layout -> A-layout into this wave's LDS.
    // Scatter offset decomposes: laneConst + (nt>>1)*512 + (nt&1)*8 + r*16
    // (in halves), so one base VGPR + immediate-offset ds_store_b16.
    _Float16* hw = hbuf + wave * 4096;
    _Float16* hp = hw + (hi * 8 + 16 * (l16 >> 3)) * 16 + ((l16 & 7) >> 1) * 2 + (l16 & 1);
#pragma unroll
    for (int nt = 0; nt < 16; ++nt) {
        float bias = b1[nt * 16 + l16];
#pragma unroll
        for (int r = 0; r < 8; ++r) {
            float hv = acc[nt][r] + bias;
            hv = hv > 0.f ? hv : 0.f;
            hp[(nt >> 1) * 512 + (nt & 1) * 8 + r * 16] = (_Float16)hv;
        }
    }
    wg_barrier_ds();   // hbuf visible (W2 slice 0 already waited above)

    // ---------------- Layer 2: [16,256] x [256,128] -> [16,128] ----------------
    const uint4* ab = (const uint4*)hw + lane * 2;   // A frags: +ks*64 [+1]

    v8f acc2[8];
#pragma unroll
    for (int i = 0; i < 8; ++i) acc2[i] = (v8f){0.f,0.f,0.f,0.f,0.f,0.f,0.f,0.f};

#pragma unroll
    for (int ks = 0; ks < 8; ++ks) {
        if (ks < 7) {
            async_slice32((ks & 1) ? l2d0 : l2d1, voff32, W2p + (ks + 1) * 2048);
        }
        Frag A;
        A.q[0] = ab[ks * 64 + 0];
        A.q[1] = ab[ks * 64 + 1];
#pragma unroll
        for (int nt = 0; nt < 8; ++nt) {
            Frag B;
            B.q[0] = wbase[(ks & 1) * 1024 + nt * 64 + 0];
            B.q[1] = wbase[(ks & 1) * 1024 + nt * 64 + 1];
            acc2[nt] = __builtin_amdgcn_wmma_f32_16x16x32_f16(
                false, A.v, false, B.v, (short)0, acc2[nt], false, false);
        }
        wg_barrier_async_ds();
    }

    // Bias + store f32 output. C layout: vgpr r, lanes0-15 -> row r, lanes16-31 -> row r+8.
    const size_t row_base = (size_t)blockIdx.x * MTILE + wave * 16 + hi * 8;
#pragma unroll
    for (int nt = 0; nt < 8; ++nt) {
        int col = nt * 16 + l16;
        float bias = b2[col];
#pragma unroll
        for (int r = 0; r < 8; ++r) {
            out[(row_base + r) * DIMO + col] = acc2[nt][r] + bias;
        }
    }
}

extern "C" void kernel_launch(void* const* d_in, const int* in_sizes, int n_in,
                              void* d_out, int out_size, void* d_ws, size_t ws_size,
                              hipStream_t stream) {
    const float* edge_attr   = (const float*)d_in[0];
    const float* node_attr   = (const float*)d_in[1];
    const float* global_attr = (const float*)d_in[2];
    const int*   senders     = (const int*)d_in[3];
    const int*   receivers   = (const int*)d_in[4];
    const float* W1          = (const float*)d_in[5];
    const float* b1          = (const float*)d_in[6];
    const float* W2          = (const float*)d_in[7];
    const float* b2          = (const float*)d_in[8];
    float*       out         = (float*)d_out;

    uint32_t* W1p = (uint32_t*)d_ws;            // 65536 dwords (256 KB)
    uint32_t* W2p = W1p + 65536;                // 16384 dwords (64 KB)
    (void)n_in; (void)out_size; (void)ws_size;

    // Pack weights to f16 B-fragment layout (81920 dwords total).
    pack_weights_kernel<<<(65536 + 16384) / 256, 256, 0, stream>>>(W1, W2, W1p, W2p);

    const int E = in_sizes[0] / DIMD;           // 640000
    edge_mlp_kernel<<<E / MTILE, 256, 0, stream>>>(
        edge_attr, node_attr, global_attr, senders, receivers,
        W1p, b1, W2p, b2, out);
}
